// GlobalAttention_53008486367940
// MI455X (gfx1250) — matrix operand order repaired and checked by reading, not compiled, not run on hardware
//
#include <hip/hip_runtime.h>
#include <hip/hip_bf16.h>
#include <stdint.h>

// Problem constants (match reference)
#define B_DIM 64
#define N_DIM 1024
#define F_DIM 128
#define D_DIM 256
// dk = D^(-0.5) = 256^-0.5
#define DK_SCALE 0.0625f

typedef __attribute__((ext_vector_type(16))) __bf16 v16bf;
typedef __attribute__((ext_vector_type(8)))  float  v8f;

union BfFrag {
    v16bf  v;
    __bf16 h[16];
    uint4  q[2];
};

// Native conversion: lowers to hardware v_cvt (packed where possible) on gfx1250.
__device__ __forceinline__ void cvt8(__bf16* dst, float4 a, float4 b) {
    dst[0] = (__bf16)a.x; dst[1] = (__bf16)a.y;
    dst[2] = (__bf16)a.z; dst[3] = (__bf16)a.w;
    dst[4] = (__bf16)b.x; dst[5] = (__bf16)b.y;
    dst[6] = (__bf16)b.z; dst[7] = (__bf16)b.w;
}

__device__ __forceinline__ v8f wmma_bf16(const BfFrag& a, const BfFrag& b, v8f c) {
    // D = A(16x32 bf16) * B(32x16 bf16) + C(16x16 f32)
    return __builtin_amdgcn_wmma_f32_16x16x32_bf16(
        /*neg_a=*/false, a.v, /*neg_b=*/false, b.v,
        /*c_mod=*/(short)0, c, /*reuse_a=*/false, /*reuse_b=*/false);
}

// ---------------------------------------------------------------------------
// Kernel 0: transpose+convert Wq/Wk/Wv (FxD fp32) -> wT (3 x D x F bf16),
//           and zero the context accumulator region of d_out.
// ---------------------------------------------------------------------------
__global__ __launch_bounds__(256) void prep_kernel(
    const float* __restrict__ Wq, const float* __restrict__ Wk,
    const float* __restrict__ Wv,
    __bf16* __restrict__ wT,           // [3][D][F] bf16
    float* __restrict__ context)       // [B*D] fp32, zeroed
{
    int t = blockIdx.x * 256 + threadIdx.x;
    const int total = 3 * D_DIM * F_DIM;
    if (t < total) {
        int m   = t / (D_DIM * F_DIM);
        int rem = t - m * (D_DIM * F_DIM);
        int d   = rem / F_DIM;
        int f   = rem - d * F_DIM;
        const float* W = (m == 0) ? Wq : (m == 1) ? Wk : Wv;
        wT[t] = (__bf16)W[(size_t)f * D_DIM + d];
    }
    if (t < B_DIM * D_DIM) context[t] = 0.0f;
}

// ---------------------------------------------------------------------------
// Kernel 1: QKV projection.  One wave per 16x16 output tile.
//   q,k -> row-major bf16 [B*N, D];  v -> transposed bf16 [B][D][N]
// ---------------------------------------------------------------------------
__global__ __launch_bounds__(256) void qkv_kernel(
    const float* __restrict__ x,            // [B*N, F] fp32
    const __bf16* __restrict__ wT,          // [3][D][F] bf16
    const float* __restrict__ bq, const float* __restrict__ bk,
    const float* __restrict__ bv,
    __bf16* __restrict__ qbuf,              // [B*N, D] bf16
    __bf16* __restrict__ kbuf,              // [B*N, D] bf16
    __bf16* __restrict__ vT)                // [B][D][N] bf16
{
    const int lane  = threadIdx.x & 31;
    const int wv    = threadIdx.x >> 5;
    const int waveG = blockIdx.x * 8 + wv;
    const int rowTile = waveG >> 4;      // (B*N)/16 tiles
    const int colTile = waveG & 15;      // D/16 tiles
    const int row0 = rowTile * 16;
    const int col0 = colTile * 16;
    const int lhalf = lane >> 4;         // 0 | 1
    const int lcol  = lane & 15;

    // A fragments: x rows (fp32 -> bf16), 4 K-chunks of 32 over F=128.
    // 16-bit A 16x32 layout: lanes 0-15 hold K = kb+0..7 (e0-7) and kb+16..23
    // (e8-15); lanes 16-31 the same shifted by +8.
    BfFrag afrag[4];
    {
        const float* xrow = x + (size_t)(row0 + lcol) * F_DIM;
        #pragma unroll
        for (int kc = 0; kc < 4; ++kc) {
            int fb = kc * 32 + lhalf * 8;
            float4 lo0 = *(const float4*)(xrow + fb);
            float4 lo1 = *(const float4*)(xrow + fb + 4);
            float4 hi0 = *(const float4*)(xrow + fb + 16);
            float4 hi1 = *(const float4*)(xrow + fb + 20);
            cvt8(&afrag[kc].h[0], lo0, lo1);
            cvt8(&afrag[kc].h[8], hi0, hi1);
        }
    }

    const float* biases[3] = { bq, bk, bv };
    const v8f vzero = {0.f,0.f,0.f,0.f,0.f,0.f,0.f,0.f};

    #pragma unroll
    for (int m = 0; m < 3; ++m) {
        // B fragment (32x16): lane = output col d, lanes 0-15 K=0..7,16..23,
        // lanes 16-31 K=8..15,24..31 -> two 16B chunks along F in wT[m][d][.]
        const __bf16* WTm =
            wT + ((size_t)m * D_DIM + (size_t)(col0 + lcol)) * F_DIM;
        v8f acc = vzero;
        #pragma unroll
        for (int kc = 0; kc < 4; ++kc) {
            const __bf16* p = WTm + kc * 32 + lhalf * 8;
            BfFrag bf;
            bf.q[0] = *(const uint4*)(p);
            bf.q[1] = *(const uint4*)(p + 16);
            acc = wmma_bf16(afrag[kc], bf, acc);
        }
        float bias = biases[m][col0 + lcol];

        if (m < 2) {
            // row-major bf16 store: lane owns column (col0+lcol), rows r+8*lhalf
            __bf16* buf = (m == 0) ? qbuf : kbuf;
            #pragma unroll
            for (int r = 0; r < 8; ++r) {
                int row = row0 + r + 8 * lhalf;
                buf[(size_t)row * D_DIM + col0 + lcol] = (__bf16)(acc[r] + bias);
            }
        } else {
            // transposed store vT[b][d][n]: lane's 8 rows are contiguous in n
            union { uint4 q; __bf16 h[8]; } pk;
            #pragma unroll
            for (int r = 0; r < 8; ++r)
                pk.h[r] = (__bf16)(acc[r] + bias);
            int bidx  = row0 / N_DIM;
            int nbase = (row0 % N_DIM) + 8 * lhalf;
            int d     = col0 + lcol;
            *(uint4*)(vT + ((size_t)bidx * D_DIM + d) * N_DIM + nbase) = pk.q;
        }
    }
}

// ---------------------------------------------------------------------------
// Kernel 2: per (b, 16-row tile): energy (WMMA) -> mask*dk -> softmax ->
//           mask -> write attn -> attn@V (WMMA) -> atomic context accumulate.
// Block = 512 threads = 16 waves.
// ---------------------------------------------------------------------------
__global__ __launch_bounds__(512) void attn_kernel(
    const __bf16* __restrict__ qbuf,          // [B*N, D] bf16
    const __bf16* __restrict__ kbuf,          // [B*N, D] bf16
    const __bf16* __restrict__ vT,            // [B][D][N] bf16
    const float* __restrict__ mask,           // [B,N,N] fp32 (0/1)
    float* __restrict__ attn_out,             // [B,N,N] fp32
    float* __restrict__ context)              // [B,D] fp32 (atomic acc)
{
    __shared__ float         sE[16][N_DIM + 4];   // scores -> exp -> attn
    __shared__ unsigned char sMask[16][N_DIM];    // binary mask

    const int lane  = threadIdx.x & 31;
    const int w     = threadIdx.x >> 5;          // wave 0..15
    const int b     = blockIdx.x >> 6;           // N/16 = 64 row tiles
    const int nTile = blockIdx.x & 63;
    const int n0    = nTile * 16;
    const int lhalf = lane >> 4;
    const int lcol  = lane & 15;
    const v8f vzero = {0.f,0.f,0.f,0.f,0.f,0.f,0.f,0.f};

    // ---- Phase 1: energy tile, rows n0..n0+15, cols [w*64, w*64+64)
    {
        const __bf16* qrow =
            qbuf + (size_t)(b * N_DIM + n0 + lcol) * D_DIM;
        v8f acc[4] = {vzero, vzero, vzero, vzero};
        #pragma unroll
        for (int kc = 0; kc < 8; ++kc) {            // K = D = 256
            BfFrag a;
            int db = kc * 32 + lhalf * 8;
            a.q[0] = *(const uint4*)(qrow + db);
            a.q[1] = *(const uint4*)(qrow + db + 16);
            #pragma unroll
            for (int sub = 0; sub < 4; ++sub) {
                int mcol = w * 64 + sub * 16 + lcol;
                const __bf16* kp =
                    kbuf + (size_t)(b * N_DIM + mcol) * D_DIM
                         + kc * 32 + lhalf * 8;
                BfFrag bb;
                bb.q[0] = *(const uint4*)(kp);
                bb.q[1] = *(const uint4*)(kp + 16);
                acc[sub] = wmma_bf16(a, bb, acc[sub]);
            }
        }
        #pragma unroll
        for (int sub = 0; sub < 4; ++sub) {
            int mcol = w * 64 + sub * 16 + lcol;
            #pragma unroll
            for (int r = 0; r < 8; ++r) {
                int rl = r + 8 * lhalf;
                float mv = mask[((size_t)(b * N_DIM + n0 + rl)) * N_DIM + mcol];
                sE[rl][mcol]    = acc[sub][r] * mv * DK_SCALE;
                sMask[rl][mcol] = (unsigned char)(mv != 0.0f);
            }
        }
    }
    __syncthreads();

    // ---- Phase 2: softmax of row w (1024 cols, 32 per lane)
    {
        float rmax = -3.402823466e38f;
        #pragma unroll 4
        for (int i = 0; i < N_DIM / 32; ++i)
            rmax = fmaxf(rmax, sE[w][i * 32 + lane]);
        #pragma unroll
        for (int off = 16; off; off >>= 1)
            rmax = fmaxf(rmax, __shfl_xor(rmax, off, 32));

        float rsum = 0.0f;
        #pragma unroll 4
        for (int i = 0; i < N_DIM / 32; ++i) {
            int c = i * 32 + lane;
            float e = __expf(sE[w][c] - rmax);
            sE[w][c] = e;
            rsum += e;
        }
        #pragma unroll
        for (int off = 16; off; off >>= 1)
            rsum += __shfl_xor(rsum, off, 32);

        float inv = 1.0f / rsum;
        size_t outRow = ((size_t)(b * N_DIM + n0 + w)) * N_DIM;
        #pragma unroll 4
        for (int i = 0; i < N_DIM / 32; ++i) {
            int c = i * 32 + lane;
            float a = sE[w][c] * inv * (float)sMask[w][c];
            sE[w][c] = a;                  // keep normalized attn for phase 3
            attn_out[outRow + c] = a;      // coalesced global store
        }
    }
    __syncthreads();

    // ---- Phase 3: P = attn(16x1024) @ V(1024x16 tile), d0 = w*16
    {
        const int d0 = w * 16;
        const __bf16* vrow =
            vT + ((size_t)b * D_DIM + d0 + lcol) * N_DIM;
        v8f acc = vzero;
        #pragma unroll 4
        for (int kc = 0; kc < 32; ++kc) {           // K = N = 1024
            BfFrag a;
            int cb = kc * 32 + lhalf * 8;
            const float* sp = &sE[lcol][cb];
            float4 lo0 = *(const float4*)(sp);
            float4 lo1 = *(const float4*)(sp + 4);
            float4 hi0 = *(const float4*)(sp + 16);
            float4 hi1 = *(const float4*)(sp + 20);
            cvt8(&a.h[0], lo0, lo1);
            cvt8(&a.h[8], hi0, hi1);
            const __bf16* vp = vrow + kc * 32 + lhalf * 8;
            BfFrag bb;
            bb.q[0] = *(const uint4*)(vp);
            bb.q[1] = *(const uint4*)(vp + 16);
            acc = wmma_bf16(a, bb, acc);
        }
        // reduce over the 16 rows of the tile: 8 per lane + partner lane
        float part = acc[0] + acc[1] + acc[2] + acc[3]
                   + acc[4] + acc[5] + acc[6] + acc[7];
        part += __shfl_xor(part, 16, 32);
        if (lane < 16)
            atomicAdd(&context[b * D_DIM + d0 + lane], part);
    }
}

// ---------------------------------------------------------------------------
extern "C" void kernel_launch(void* const* d_in, const int* in_sizes, int n_in,
                              void* d_out, int out_size, void* d_ws, size_t ws_size,
                              hipStream_t stream) {
    (void)in_sizes; (void)n_in; (void)out_size; (void)ws_size;
    const float* x    = (const float*)d_in[0];  // atom_query [B,N,F]
    const float* mask = (const float*)d_in[1];  // [B,N,N]
    const float* Wq   = (const float*)d_in[2];
    const float* bq   = (const float*)d_in[3];
    const float* Wk   = (const float*)d_in[4];
    const float* bk   = (const float*)d_in[5];
    const float* Wv   = (const float*)d_in[6];
    const float* bv   = (const float*)d_in[7];

    float* attn    = (float*)d_out;                              // [B,N,N]
    float* context = attn + (size_t)B_DIM * N_DIM * N_DIM;       // [B,D]

    // workspace layout (bf16): wT[3*D*F] | q[BN*D] | k[BN*D] | vT[B*D*N]
    __bf16* wT = (__bf16*)d_ws;
    __bf16* qb = wT + (size_t)3 * D_DIM * F_DIM;
    __bf16* kb = qb + (size_t)B_DIM * N_DIM * D_DIM;
    __bf16* vt = kb + (size_t)B_DIM * N_DIM * D_DIM;

    prep_kernel<<<384, 256, 0, stream>>>(Wq, Wk, Wv, wT, context);

    // (B*N/16) * (D/16) wave tiles, 8 waves per 256-thread block
    qkv_kernel<<<(B_DIM * N_DIM / 16) * (D_DIM / 16) / 8, 256, 0, stream>>>(
        x, wT, bq, bk, bv, qb, kb, vt);

    // one 16-wave block per (b, 16-row tile)
    attn_kernel<<<B_DIM * (N_DIM / 16), 512, 0, stream>>>(
        qb, kb, vt, mask, attn, context);
}
